// MackeyGlassSNN_15831249453627
// MI455X (gfx1250) — 8-bit matrix operand order repaired and checked by reading, compiled, not run on hardware
//
#include <hip/hip_runtime.h>

#define BATCH   1024
#define T_STEPS 4096
#define N_IN    8
#define N_HID   256
#define NTHREADS 512

typedef __attribute__((ext_vector_type(16))) _Float16 v16h;
typedef __attribute__((ext_vector_type(8)))  _Float16 v8h;
typedef __attribute__((ext_vector_type(8)))  float    v8f;
typedef __attribute__((ext_vector_type(8)))  int      v8i;
typedef __attribute__((ext_vector_type(2)))  int      v2i;

#define SB_STRIDE 272
#define XC_STRIDE 520

__global__ __launch_bounds__(NTHREADS, 1)
void adlif_snn_kernel(const float* __restrict__ x,
                      const float* __restrict__ W1,
                      const float* __restrict__ Wrec,
                      const float* __restrict__ W2,
                      const float* __restrict__ alpha,
                      const float* __restrict__ rho,
                      const float* __restrict__ beta_a,
                      float* __restrict__ out)
{
    __shared__ __align__(16) unsigned char sbuf[2][16][SB_STRIDE];
    __shared__ __align__(16) _Float16 xch[16][XC_STRIDE];
    __shared__ float accum[16];

    const int tid  = threadIdx.x;
    const int lane = tid & 31;
    const int wave = tid >> 5;
    const int col  = lane & 15;
    const int hi   = lane >> 4;
    const int b0   = blockIdx.x * 16;

    for (int i = tid; i < 2 * 16 * SB_STRIDE; i += NTHREADS)
        ((unsigned char*)sbuf)[i] = 0;
    if (tid < 16) accum[tid] = 0.f;

    const int n = wave * 16 + col;
    const float alp = alpha[n], oma = 1.f - alp;
    const float rh  = rho[n],   omr = 1.f - rh;
    const float bet = beta_a[n];
    const float w2v = W2[n];

    v8i bfrag[4];
    float scl;
    {
        const float* wr = Wrec + (size_t)n * N_HID;
        float mx = 0.f;
        for (int k = 0; k < N_HID; ++k) mx = fmaxf(mx, fabsf(wr[k]));
        const float qs = (mx > 0.f) ? (127.f / mx) : 0.f;
        scl = (mx > 0.f) ? (mx / 127.f) : 0.f;

#pragma unroll
        for (int c = 0; c < 4; ++c) {
            v8i b;
#pragma unroll
            for (int v = 0; v < 8; ++v) {
                const int k0 = 64 * c + ((v < 4) ? (16 * hi + 4 * v)
                                                 : (32 + 16 * hi + 4 * (v - 4)));
                unsigned wrd = 0;
#pragma unroll
                for (int e = 0; e < 4; ++e) {
                    int qi = (int)rintf(wr[k0 + e] * qs);
                    qi = qi > 127 ? 127 : (qi < -127 ? -127 : qi);
                    wrd |= ((unsigned)(qi & 0xff)) << (8 * e);
                }
                b[v] = (int)wrd;
            }
            bfrag[c] = b;
        }
    }

    v16h w1b = {};
    if (!hi) {
#pragma unroll
        for (int e = 0; e < 8; ++e) w1b[e] = (_Float16)W1[n * N_IN + e];
    }

    auto stage_x = [&](int t0) {
        for (int i = tid; i < 2048; i += NTHREADS) {
            const int fi  = i * 4;
            const int m   = fi >> 9;
            const int rem = fi & 511;
            const float* src =
                x + ((size_t)(b0 + m) * T_STEPS + t0) * N_IN + rem;
            const float4 vv = *(const float4*)src;
            _Float16* dst = &xch[m][rem];
            dst[0] = (_Float16)vv.x; dst[1] = (_Float16)vv.y;
            dst[2] = (_Float16)vv.z; dst[3] = (_Float16)vv.w;
            if (t0 + 64 < T_STEPS)
                __builtin_prefetch(src + 64 * N_IN, 0, 1);
        }
    };

    float vmem[8], amem[8], sprev[8];
#pragma unroll
    for (int r = 0; r < 8; ++r) { vmem[r] = 0.f; amem[r] = 0.f; sprev[r] = 0.f; }

    stage_x(0);
    __syncthreads();

    const int mrow = 8 * hi;

    for (int t = 0; t < T_STEPS; ++t) {
        const int buf  = t & 1;
        const int nxt  = buf ^ 1;
        const int tloc = t & 63;

        v16h xa = {};
        if (!hi) {
            const v8h xv = *(const v8h*)&xch[col][tloc * 8];
#pragma unroll
            for (int e = 0; e < 8; ++e) xa[e] = xv[e];
        }
        v8f fz = {};
        v8f f0 = __builtin_amdgcn_wmma_f32_16x16x32_f16(
            false, xa, false, w1b, (short)0, fz, false, false);

        v8i i0 = {};
        const unsigned char* srow = &sbuf[buf][col][0];
#pragma unroll
        for (int c = 0; c < 4; ++c) {
            v8i af;
#pragma unroll
            for (int q = 0; q < 4; ++q) {
                const v2i seg =
                    *(const v2i*)(srow + 64 * c + 16 * hi + 32 * (q >> 1) + 8 * (q & 1));
                af[2 * q]     = seg.x;
                af[2 * q + 1] = seg.y;
            }
            i0 = __builtin_amdgcn_wmma_i32_16x16x64_iu8(
                false, af, true, bfrag[c], i0, false, false);
        }

#pragma unroll
        for (int r = 0; r < 8; ++r) {
            const float I  = (float)i0[r] * scl + f0[r];
            const float an = rh * amem[r] + omr * vmem[r] + bet * sprev[r];
            float vn = alp * vmem[r] + oma * (I - an);
            const bool  spk = (vn >= 1.f);
            const float sf  = spk ? 1.f : 0.f;
            vmem[r]  = vn - sf;
            amem[r]  = an;
            sprev[r] = sf;
            sbuf[nxt][mrow + r][n] = spk ? 1 : 0;
        }

        __syncthreads();

        if (((t + 1) & 63) == 0 && (t + 1) < T_STEPS) {
            stage_x(t + 1);
            __syncthreads();
        }
    }

#pragma unroll
    for (int r = 0; r < 8; ++r)
        atomicAdd(&accum[mrow + r], vmem[r] * w2v);
    __syncthreads();
    if (tid < 16) out[b0 + tid] = accum[tid];
}

extern "C" void kernel_launch(void* const* d_in, const int* in_sizes, int n_in,
                              void* d_out, int out_size, void* d_ws,
                              size_t ws_size, hipStream_t stream)
{
    (void)in_sizes; (void)n_in; (void)out_size; (void)d_ws; (void)ws_size;
    const float* x     = (const float*)d_in[0];
    const float* W1    = (const float*)d_in[1];
    const float* Wrec  = (const float*)d_in[2];
    const float* W2    = (const float*)d_in[3];
    const float* alpha = (const float*)d_in[4];
    const float* rho   = (const float*)d_in[5];
    const float* beta  = (const float*)d_in[6];
    float* out = (float*)d_out;

    adlif_snn_kernel<<<BATCH / 16, NTHREADS, 0, stream>>>(
        x, W1, Wrec, W2, alpha, rho, beta, out);
}
